// TasNetEncoder_82214263980409
// MI455X (gfx1250) — matrix operand order repaired and checked by reading, not compiled, not run on hardware
//
#include <hip/hip_runtime.h>

// ---------------- problem constants (from reference) ----------------
#define W_TAP   4
#define OUT_CH  256
#define OUT_F   128
#define B_SZ    16
#define T_LEN   64000
#define L_LEN   32000          // ceil(2*T/W)
#define NT      64             // columns (l) per workgroup
#define RS      264            // h_lds row stride in bf16 (256 + 8 pad -> bank-conflict-free)

typedef __attribute__((ext_vector_type(2)))  float  v2f;
typedef __attribute__((ext_vector_type(8)))  float  v8f;
typedef __attribute__((ext_vector_type(8)))  __bf16 v8bf;
typedef __attribute__((ext_vector_type(16))) __bf16 v16bf;

union V16BF { v16bf v; v8bf h[2]; };

// single-instruction relu: v_med3_f32(x, 0, +inf); inputs are NaN-free
__device__ __forceinline__ float relu1(float v) {
#if __has_builtin(__builtin_amdgcn_fmed3f)
    return __builtin_amdgcn_fmed3f(v, 0.0f, __builtin_inff());
#else
    return __builtin_fmaxf(v, 0.0f);
#endif
}

// ---------------- prep: lin_w f32 -> bf16 in workspace; padding ints ----------------
__global__ void tasnet_prep_kernel(const float* __restrict__ lin_w,
                                   __bf16* __restrict__ lin_bf,
                                   int* __restrict__ pad_out) {
    int i = blockIdx.x * 256 + threadIdx.x;
    if (i < OUT_F * OUT_CH) lin_bf[i] = (__bf16)lin_w[i];
    if (i == 0) { pad_out[0] = 2; pad_out[1] = 0; }   // (pad_l, pad_r) int32
}

// ---------------- fused encoder ----------------
__global__ __launch_bounds__(256)
void tasnet_enc_kernel(const float* __restrict__ x,
                       const float* __restrict__ conv_w,
                       const __bf16* __restrict__ lin_bf,
                       float* __restrict__ out,
                       float* __restrict__ frames_out) {
    // frames packed as K-pairs: fr_lds[(p*NT + n)*2 + c] = frames[2p+c][n]
    __shared__ __align__(16) float  fr_lds[2 * NT * 2];
    // h (post-relu, bf16), column-major over n, K contiguous: h_lds[n*RS + e]
    __shared__ __align__(16) __bf16 h_lds[NT * RS];

    const int tid  = threadIdx.x;
    const int lane = tid & 31;
    const int wv   = tid >> 5;      // 8 waves
    const int hi   = lane >> 4;     // half-wave: 0/1
    const int lm   = lane & 15;

    const int chunk = blockIdx.x;
    const int b  = chunk / (L_LEN / NT);
    const int l0 = (chunk % (L_LEN / NT)) * NT;

    // ---- Phase 1: build frame tile (4 x NT) + write frames output ----
    {
        int w_idx = tid >> 6;                 // 0..3
        int n     = tid & 63;                 // 0..63
        int l     = l0 + n;
        int src   = 2 * l + w_idx - 2;        // pad_l = 2
        float val = (src >= 0 && src < T_LEN) ? x[(size_t)b * T_LEN + src] : 0.0f;
        fr_lds[((w_idx >> 1) * NT + n) * 2 + (w_idx & 1)] = val;
        frames_out[(size_t)b * (W_TAP * L_LEN) + (size_t)w_idx * L_LEN + l] = val;
    }
    __syncthreads();

    // ---- Phase 2: conv (f32 WMMA, K=4 exact) + relu -> packed bf16 h in LDS ----
    // wave wv handles channel tiles mt = 2*wv, 2*wv+1; all 4 N-tiles
#pragma unroll
    for (int i = 0; i < 2; ++i) {
        const int mt = wv * 2 + i;            // 16-channel tile, 0..15
        // A 16x4 f32: lane holds row lm, K = hi*2, hi*2+1
        v2f a = *(const v2f*)(conv_w + (mt * 16 + lm) * W_TAP + hi * 2);
#pragma unroll
        for (int nt = 0; nt < 4; ++nt) {
            // B 4x16 f32: lane holds col lm of N-tile, K = hi*2, hi*2+1
            v2f bb = *(const v2f*)(fr_lds + (hi * NT + nt * 16 + lm) * 2);
            v8f c = {};
#if __has_builtin(__builtin_amdgcn_wmma_f32_16x16x4_f32)
            c = __builtin_amdgcn_wmma_f32_16x16x4_f32(false, a, false, bb,
                                                      (short)0, c, false, false);
#else
            // layout-identical VALU fallback: C elem r <-> M = r + hi*8, N = lm
#pragma unroll
            for (int r = 0; r < 8; ++r) {
                int e = mt * 16 + r + hi * 8;
                float acc = 0.0f;
#pragma unroll
                for (int k = 0; k < W_TAP; ++k)
                    acc += conv_w[e * W_TAP + k] *
                           fr_lds[((k >> 1) * NT + nt * 16 + lm) * 2 + (k & 1)];
                c[r] = acc;
            }
#endif
            // C elem r maps to channel e = mt*16 + r + hi*8: the 8 values per lane
            // are CONTIGUOUS channels -> relu, pack to v8bf, one ds_store_b128.
            v8bf hv;
#pragma unroll
            for (int r = 0; r < 8; ++r)
                hv[r] = (__bf16)relu1(c[r]);
            const int n = nt * 16 + lm;
            *(v8bf*)(h_lds + n * RS + mt * 16 + hi * 8) = hv;
        }
    }
    __syncthreads();

    // ---- Phase 3: out(128 x NT) = lin(128x256) @ h(256 x NT), bf16 WMMA ----
    v8f acc[4] = {{}, {}, {}, {}};
    const int m = wv * 16 + lm;               // wave wv owns M-tile wv
#pragma unroll
    for (int kb = 0; kb < OUT_CH; kb += 32) {
        // A 16x32 bf16: lane row m; elems 0-7 = K kb+hi*8.., elems 8-15 = K kb+16+hi*8..
        V16BF a;
        const __bf16* ap = lin_bf + (size_t)m * OUT_CH + kb + hi * 8;
        a.h[0] = *(const v8bf*)(ap);
        a.h[1] = *(const v8bf*)(ap + 16);
#pragma unroll
        for (int nt = 0; nt < 4; ++nt) {
            // B 32x16 bf16: lane col lm; elems 0-7 = K kb+hi*8.., elems 8-15 = K kb+16+hi*8..
            V16BF bb;
            const __bf16* bp = h_lds + (nt * 16 + lm) * RS + kb + hi * 8;
            bb.h[0] = *(const v8bf*)(bp);
            bb.h[1] = *(const v8bf*)(bp + 16);
            acc[nt] = __builtin_amdgcn_wmma_f32_16x16x32_bf16(
                          false, a.v, false, bb.v, (short)0, acc[nt], false, false);
        }
    }

    // ---- Phase 4: store out (coalesced over l) ----
#pragma unroll
    for (int nt = 0; nt < 4; ++nt) {
        const int l = l0 + nt * 16 + lm;
#pragma unroll
        for (int r = 0; r < 8; ++r) {
            const int f = wv * 16 + r + hi * 8;
            out[(size_t)b * (OUT_F * L_LEN) + (size_t)f * L_LEN + l] = acc[nt][r];
        }
    }
}

// ---------------- launch ----------------
extern "C" void kernel_launch(void* const* d_in, const int* in_sizes, int n_in,
                              void* d_out, int out_size, void* d_ws, size_t ws_size,
                              hipStream_t stream) {
    const float* x      = (const float*)d_in[0];   // (16, 64000)
    const float* conv_w = (const float*)d_in[1];   // (256, 4)
    const float* lin_w  = (const float*)d_in[2];   // (128, 256)

    float* out        = (float*)d_out;                                   // (16,128,32000)
    float* frames_out = out + (size_t)B_SZ * OUT_F * L_LEN;              // (16,4,32000)
    int*   pad_out    = (int*)(frames_out + (size_t)B_SZ * W_TAP * L_LEN); // 2 x int32

    __bf16* lin_bf = (__bf16*)d_ws;   // 64 KB

    tasnet_prep_kernel<<<(OUT_F * OUT_CH + 255) / 256, 256, 0, stream>>>(
        lin_w, lin_bf, pad_out);

    tasnet_enc_kernel<<<B_SZ * (L_LEN / NT), 256, 0, stream>>>(
        x, conv_w, lin_bf, out, frames_out);
}
